// TensorProductConvLayer_19894288515355
// MI455X (gfx1250) — matrix operand order repaired and checked by reading, not compiled, not run on hardware
//
#include <hip/hip_runtime.h>
#include <stdint.h>

// ---------------- constants ----------------
#define NNODES   10000
#define NEDGES   160000
#define INV_S3   0.57735026918962576f     // 1/sqrt(3)
#define ALPHA0   0.17677669529663687f     // sqrt(1/32)
#define ALPHA1   0.30618621784789724f     // sqrt(3/32)
#define EPS_BN   1e-5f

typedef __attribute__((ext_vector_type(16))) __bf16 v16bf;
typedef __attribute__((ext_vector_type(8)))  float  v8f;

union BFrag {                 // 16 bf16 = one WMMA A/B operand (8 VGPRs)
  v16bf          v;
  unsigned short u[16];
  uint4          q[2];
};

static __device__ __forceinline__ unsigned short f2bf(float f) {
  unsigned int x = __float_as_uint(f);
  unsigned int r = x + 0x7FFFu + ((x >> 16) & 1u);   // round-to-nearest-even
  return (unsigned short)(r >> 16);
}

// ---------------- kernel 0: zero workspace + pack W1/W2 into bf16 fragment order ----
// B-matrix (K=32 step) per-lane layout: ncol = tile*16 + (lane&15),
// K = kstep*32 + 8*(lane>>4) + (slot&7) + 16*(slot>>3)   (slot = 0..15)
__global__ __launch_bounds__(256)
void prep_kernel(float* __restrict__ zero_base, int nz,
                 const float* __restrict__ W1, const float* __restrict__ W2,
                 unsigned short* __restrict__ w1p, unsigned short* __restrict__ w2p) {
  int i = blockIdx.x * 256 + threadIdx.x;
  if (i < nz) zero_base[i] = 0.0f;
  if (i < 4096) {           // W1: 4 tiles * 2 ksteps * 32 lanes * 16 slots
    int slot = i & 15, L = (i >> 4) & 31, ks = (i >> 9) & 1, t = i >> 10;
    int K = ks * 32 + ((L >> 4) << 3) + (slot & 7) + ((slot >> 3) << 4);
    int col = (t << 4) + (L & 15);
    w1p[i] = f2bf(W1[K * 64 + col]);
  }
  if (i < 65536) {          // W2: 64 tiles * 2 * 32 * 16
    int slot = i & 15, L = (i >> 4) & 31, ks = (i >> 9) & 1, t = i >> 10;
    int K = ks * 32 + ((L >> 4) << 3) + (slot & 7) + ((slot >> 3) << 4);
    int col = (t << 4) + (L & 15);
    w2p[i] = f2bf(W2[K * 1024 + col]);
  }
}

// ---------------- kernel 1: fused edge MLP (WMMA) + tensor product + scatter ------
struct alignas(16) WaveShared {
  unsigned short hT[16][64];   // h tile (bf16), row = edge-in-tile, col = k
  float xs[16][16];            // [edge][u]
  float xv[3][16][16];         // [i][edge][u]
  float a3[16][16];            // xv . shv  [edge][u]
  float shs[16];
  float shv[3][16];
  int   dst[16];
};                             // 7488 B

__global__ __launch_bounds__(256)
void edge_kernel(const float* __restrict__ node_attr,
                 const int*   __restrict__ edge_index,
                 const float* __restrict__ edge_attr,
                 const float* __restrict__ edge_sh,
                 const float* __restrict__ b1v,
                 const float* __restrict__ b2v,
                 const unsigned short* __restrict__ w1p,
                 const unsigned short* __restrict__ w2p,
                 float* __restrict__ seg,
                 float* __restrict__ cnt) {
  __shared__ WaveShared sm[8];
  const int wv   = threadIdx.x >> 5;
  const int lane = threadIdx.x & 31;
  const int e0   = (blockIdx.x * 8 + wv) * 16;      // grid sized exactly: 1250*8*16 = 160000
  WaveShared& S  = sm[wv];

  // ---- phase 1: lanes 0..15 stage TP coefficients for the 16 edges ----
  if (lane < 16) {
    const int e    = e0 + lane;
    const int srci = edge_index[e];
    const int dsti = edge_index[NEDGES + e];
    const float* na = node_attr + (size_t)srci * 64;
    const float* sh = edge_sh   + (size_t)e * 4;
    const float s0 = sh[0], s1 = sh[1], s2 = sh[2], s3 = sh[3];
    S.shs[lane] = s0; S.shv[0][lane] = s1; S.shv[1][lane] = s2; S.shv[2][lane] = s3;
    S.dst[lane] = dsti;
#pragma unroll
    for (int u = 0; u < 16; ++u) {
      S.xs[lane][u] = na[u];
      const float x0 = na[16 + u * 3 + 0];
      const float x1 = na[16 + u * 3 + 1];
      const float x2 = na[16 + u * 3 + 2];
      S.xv[0][lane][u] = x0; S.xv[1][lane][u] = x1; S.xv[2][lane][u] = x2;
      S.a3[lane][u] = x0 * s1 + x1 * s2 + x2 * s3;
    }
    atomicAdd(&cnt[dsti], 1.0f);
  }
  __syncthreads();

  const int row = lane & 15;        // A row (edge in tile) / C column for this lane
  const int h16 = lane >> 4;        // lane half

  // ---- phase 2: edge_attr A fragments (bf16) straight from global ----
  // A layout: K = kstep*32 + e + 8*(e/8) + 8*h16  -> two contiguous runs of 8
  BFrag eaF[2];
  const float* eaRow = edge_attr + (size_t)(e0 + row) * 64;
#pragma unroll
  for (int ks = 0; ks < 2; ++ks) {
    const int o1 = ks * 32 + h16 * 8;
#pragma unroll
    for (int q = 0; q < 8; ++q) eaF[ks].u[q]     = f2bf(eaRow[o1 + q]);
#pragma unroll
    for (int q = 0; q < 8; ++q) eaF[ks].u[8 + q] = f2bf(eaRow[o1 + 16 + q]);
  }

  // ---- phase 3: GEMM1  h = relu(ea @ W1 + b1), h -> LDS (bf16) ----
  const uint4* w1q = (const uint4*)w1p;
#pragma unroll
  for (int n = 0; n < 4; ++n) {
    BFrag bf0, bf1;
    const int f0 = (n * 2 + 0) * 32 + lane;
    const int f1 = (n * 2 + 1) * 32 + lane;
    bf0.q[0] = w1q[f0 * 2]; bf0.q[1] = w1q[f0 * 2 + 1];
    bf1.q[0] = w1q[f1 * 2]; bf1.q[1] = w1q[f1 * 2 + 1];
    v8f c = {0.f, 0.f, 0.f, 0.f, 0.f, 0.f, 0.f, 0.f};
    c = __builtin_amdgcn_wmma_f32_16x16x32_bf16(false, eaF[0].v, false, bf0.v, (short)0, c, false, false);
    c = __builtin_amdgcn_wmma_f32_16x16x32_bf16(false, eaF[1].v, false, bf1.v, (short)0, c, false, false);
    const int   col  = n * 16 + row;
    const float bias = b1v[col];
#pragma unroll
    for (int r = 0; r < 8; ++r) {               // C layout: VGPR r -> row r + 8*h16, col = lane&15
      const int rm = r + h16 * 8;
      S.hT[rm][col] = f2bf(fmaxf(c[r] + bias, 0.0f));
    }
  }

  // ---- phase 4: reload h as A fragments (same-wave LDS, in-order) ----
  BFrag hA[2];
#pragma unroll
  for (int ks = 0; ks < 2; ++ks) {
    const int o1 = ks * 32 + h16 * 8;
    hA[ks].q[0] = *(const uint4*)&S.hT[row][o1];
    hA[ks].q[1] = *(const uint4*)&S.hT[row][o1 + 16];
  }

  // per-row spherical-harmonic constants
  float rs[8], rv0[8], rv1[8], rv2[8];
#pragma unroll
  for (int r = 0; r < 8; ++r) {
    const int rm = r + h16 * 8;
    rs[r] = S.shs[rm]; rv0[r] = S.shv[0][rm]; rv1[r] = S.shv[1][rm]; rv2[r] = S.shv[2][rm];
  }
  float sAcc[8] = {0, 0, 0, 0, 0, 0, 0, 0};
  float vA0[8]  = {0, 0, 0, 0, 0, 0, 0, 0};
  float vA1[8]  = {0, 0, 0, 0, 0, 0, 0, 0};
  float vA2[8]  = {0, 0, 0, 0, 0, 0, 0, 0};

  // ---- phase 5: GEMM2 tile loop fused with the tensor product ----
  // column block t = j*16+u covers w[:, j, u, 0..15]
  const uint4* w2q = (const uint4*)w2p;
  for (int u = 0; u < 16; ++u) {
    float cxs[8], ca3[8], cv0[8], cv1[8], cv2[8];
#pragma unroll
    for (int r = 0; r < 8; ++r) {
      const int rm = r + h16 * 8;
      cxs[r] = S.xs[rm][u]; ca3[r] = S.a3[rm][u];
      cv0[r] = S.xv[0][rm][u]; cv1[r] = S.xv[1][rm][u]; cv2[r] = S.xv[2][rm][u];
    }
#pragma unroll
    for (int j = 0; j < 4; ++j) {
      const int t  = j * 16 + u;
      BFrag bf0, bf1;
      const int f0 = (t * 2 + 0) * 32 + lane;
      const int f1 = (t * 2 + 1) * 32 + lane;
      bf0.q[0] = w2q[f0 * 2]; bf0.q[1] = w2q[f0 * 2 + 1];
      bf1.q[0] = w2q[f1 * 2]; bf1.q[1] = w2q[f1 * 2 + 1];
      v8f c = {0.f, 0.f, 0.f, 0.f, 0.f, 0.f, 0.f, 0.f};
      c = __builtin_amdgcn_wmma_f32_16x16x32_bf16(false, hA[0].v, false, bf0.v, (short)0, c, false, false);
      c = __builtin_amdgcn_wmma_f32_16x16x32_bf16(false, hA[1].v, false, bf1.v, (short)0, c, false, false);
      const float bias = b2v[t * 16 + row];
#pragma unroll
      for (int r = 0; r < 8; ++r) {
        const float wvl = c[r] + bias;
        if (j == 0) {
          sAcc[r] += cxs[r] * rs[r] * wvl;                       // xs*shs*w0
        } else if (j == 1) {
          const float tmp = cxs[r] * wvl;                        // xs*shv_i*w1
          vA0[r] += rv0[r] * tmp; vA1[r] += rv1[r] * tmp; vA2[r] += rv2[r] * tmp;
        } else if (j == 2) {
          const float tmp = rs[r] * wvl;                         // xv_i*shs*w2
          vA0[r] += cv0[r] * tmp; vA1[r] += cv1[r] * tmp; vA2[r] += cv2[r] * tmp;
        } else {
          sAcc[r] += INV_S3 * ca3[r] * wvl;                      // (xv.shv)*w3
        }
      }
    }
  }

  // ---- phase 6: scatter summand rows into segment sums ----
  const float SV = ALPHA1 * INV_S3;
#pragma unroll
  for (int r = 0; r < 8; ++r) {
    const int rm = r + h16 * 8;
    float* segrow = seg + (size_t)S.dst[rm] * 64;
    atomicAdd(&segrow[row],                ALPHA0 * sAcc[r]);
    atomicAdd(&segrow[16 + row * 3 + 0],   SV * vA0[r]);
    atomicAdd(&segrow[16 + row * 3 + 1],   SV * vA1[r]);
    atomicAdd(&segrow[16 + row * 3 + 2],   SV * vA2[r]);
  }
}

// ---------------- kernel 2: out = seg/max(cnt,1)+node_attr ; accumulate BN stats ----
__global__ __launch_bounds__(256)
void node_reduce_kernel(const float* __restrict__ seg, const float* __restrict__ cnt,
                        const float* __restrict__ node_attr,
                        float* __restrict__ outbuf, float* __restrict__ stats) {
  const int n = blockIdx.x * 256 + threadIdx.x;
  float ps[16], pss[16], pv[16];
#pragma unroll
  for (int u = 0; u < 16; ++u) { ps[u] = 0.f; pss[u] = 0.f; pv[u] = 0.f; }
  if (n < NNODES) {
    const float inv = 1.0f / fmaxf(cnt[n], 1.0f);
    const float* sg = seg + (size_t)n * 64;
    const float* na = node_attr + (size_t)n * 64;
    float* ob = outbuf + (size_t)n * 64;
    float rowv[64];
#pragma unroll
    for (int k = 0; k < 64; ++k) { rowv[k] = sg[k] * inv + na[k]; ob[k] = rowv[k]; }
#pragma unroll
    for (int u = 0; u < 16; ++u) {
      const float f = rowv[u];
      ps[u] = f; pss[u] = f * f;
      const float f0 = rowv[16 + 3 * u], f1 = rowv[17 + 3 * u], f2 = rowv[18 + 3 * u];
      pv[u] = (f0 * f0 + f1 * f1 + f2 * f2) * (1.0f / 3.0f);
    }
  }
  __shared__ float red[256][48];
#pragma unroll
  for (int u = 0; u < 16; ++u) {
    red[threadIdx.x][u]      = ps[u];
    red[threadIdx.x][16 + u] = pss[u];
    red[threadIdx.x][32 + u] = pv[u];
  }
  __syncthreads();
  if (threadIdx.x < 48) {
    float s = 0.f;
    for (int i = 0; i < 256; ++i) s += red[i][threadIdx.x];
    atomicAdd(&stats[threadIdx.x], s);
  }
}

// ---------------- kernel 3: batch-norm apply -> d_out ----
__global__ __launch_bounds__(256)
void bn_kernel(const float* __restrict__ outbuf, const float* __restrict__ stats,
               const float* __restrict__ bn_ws, const float* __restrict__ bn_wv,
               const float* __restrict__ bn_b, float* __restrict__ out) {
  const int n = blockIdx.x * 256 + threadIdx.x;
  if (n >= NNODES) return;
  const float invN = 1.0f / (float)NNODES;
  const float* r = outbuf + (size_t)n * 64;
  float* o = out + (size_t)n * 64;
#pragma unroll
  for (int u = 0; u < 16; ++u) {
    const float mean = stats[u] * invN;
    const float var  = stats[16 + u] * invN - mean * mean;   // ns = biased variance
    const float sc   = rsqrtf(var + EPS_BN) * bn_ws[u];
    o[u] = (r[u] - mean) * sc + bn_b[u];
    const float nv  = stats[32 + u] * invN;
    const float scv = rsqrtf(nv + EPS_BN) * bn_wv[u];
    o[16 + 3 * u + 0] = r[16 + 3 * u + 0] * scv;
    o[16 + 3 * u + 1] = r[16 + 3 * u + 1] * scv;
    o[16 + 3 * u + 2] = r[16 + 3 * u + 2] * scv;
  }
}

// ---------------- host launch ----------------
extern "C" void kernel_launch(void* const* d_in, const int* in_sizes, int n_in,
                              void* d_out, int out_size, void* d_ws, size_t ws_size,
                              hipStream_t stream) {
  (void)in_sizes; (void)n_in; (void)out_size; (void)ws_size;
  const float* node_attr = (const float*)d_in[0];
  const int*   edge_idx  = (const int*)  d_in[1];
  const float* edge_attr = (const float*)d_in[2];
  const float* edge_sh   = (const float*)d_in[3];
  const float* W1        = (const float*)d_in[4];
  const float* b1        = (const float*)d_in[5];
  const float* W2        = (const float*)d_in[6];
  const float* b2        = (const float*)d_in[7];
  const float* bn_ws     = (const float*)d_in[8];
  const float* bn_wv     = (const float*)d_in[9];
  const float* bn_b      = (const float*)d_in[10];
  float* out = (float*)d_out;

  // workspace layout
  char*  ws    = (char*)d_ws;
  float* seg   = (float*)ws;            // 640000 f32 (zeroed)
  float* cnt   = seg + 640000;          // 10000 f32  (zeroed)
  float* stats = cnt + 10000;           // 48 f32     (zeroed)
  size_t off   = (((size_t)(640000 + 10000 + 48) * 4) + 255) & ~(size_t)255;
  float* outbuf = (float*)(ws + off);   // 640000 f32
  off += (size_t)640000 * 4;
  unsigned short* w1p = (unsigned short*)(ws + off); off += 8192;
  unsigned short* w2p = (unsigned short*)(ws + off);

  const int nz = 640000 + 10000 + 48;
  prep_kernel<<<(nz + 255) / 256, 256, 0, stream>>>(seg, nz, W1, W2, w1p, w2p);
  edge_kernel<<<NEDGES / (16 * 8), 256, 0, stream>>>(node_attr, edge_idx, edge_attr,
                                                     edge_sh, b1, b2, w1p, w2p, seg, cnt);
  node_reduce_kernel<<<(NNODES + 255) / 256, 256, 0, stream>>>(seg, cnt, node_attr, outbuf, stats);
  bn_kernel<<<(NNODES + 255) / 256, 256, 0, stream>>>(outbuf, stats, bn_ws, bn_wv, bn_b, out);
}
